// CAMERAMOE_39737037422751
// MI455X (gfx1250) — matrix operand order repaired and checked by reading, not compiled, not run on hardware
//
#include <hip/hip_runtime.h>
#include <hip/hip_bf16.h>
#include <cstdint>
#include <cstddef>

// Problem constants (from reference)
#define T_TOK  8192     // B*S
#define DIN    1280
#define DHID   2048
#define NE     8
#define TILE_M 32       // tokens per FFN block (2 x 16-row WMMA tiles)
#define MT     2        // M-tiles per block
#define THREADS 256
#define NWAVES 8
#define LN_EPS 1e-6f

typedef __attribute__((ext_vector_type(16))) __bf16          v16bf;
typedef __attribute__((ext_vector_type(16))) unsigned short  v16u;
typedef __attribute__((ext_vector_type(8)))  float           v8f;
typedef __attribute__((ext_vector_type(4)))  unsigned int    v4ui;
typedef __attribute__((ext_vector_type(4)))  unsigned int    v4u;
typedef __attribute__((ext_vector_type(8)))  int             v8i;
typedef __attribute__((ext_vector_type(4)))  int             v4i;

__device__ __forceinline__ unsigned short f2bf(float f) {
  unsigned u = __float_as_uint(f);
  u += 0x7FFFu + ((u >> 16) & 1u);       // round-to-nearest-even
  return (unsigned short)(u >> 16);
}

union AFrag { v4ui q[2]; v16u s; };

// A fragment (16x32 bf16, k = 16*hi + s, same order as B): lane(hi,ln) = two contiguous 16B runs.
__device__ __forceinline__ v16u load_afrag(const unsigned short* base, int row,
                                           int stride, int k0, int hi) {
  AFrag u;
  const unsigned short* p = base + (size_t)row * stride + k0 + hi * 16;
  u.q[0] = *(const v4ui*)(p);
  u.q[1] = *(const v4ui*)(p + 8);
  return u.s;
}

// B fragment from K-major (transposed) bf16 weights: one 32B contiguous run.
__device__ __forceinline__ v16u load_bfrag(const unsigned short* wt, int n,
                                           int K, int k0, int hi) {
  return *(const v16u*)(wt + (size_t)n * K + k0 + hi * 16);
}

__device__ __forceinline__ v8f wmma_bf16(v16u a, v16u b, v8f c) {
  return __builtin_amdgcn_wmma_f32_16x16x32_bf16(
      false, __builtin_bit_cast(v16bf, a),
      false, __builtin_bit_cast(v16bf, b),
      (short)0, c, false, false);
}

// ---------------------------------------------------------------- kernel 1
__global__ void moe_zero_init(float* __restrict__ out, int n, int* __restrict__ counts) {
  int i = blockIdx.x * blockDim.x + threadIdx.x;
  if (i < n) out[i] = 0.0f;
  if (blockIdx.x == 0 && threadIdx.x < NE) counts[threadIdx.x] = 0;
}

// ---------------------------------------------------------------- kernel 2
__global__ void moe_cvt_x(const float* __restrict__ x, unsigned short* __restrict__ xb, int n) {
  int i = blockIdx.x * blockDim.x + threadIdx.x;
  if (i < n) xb[i] = f2bf(x[i]);
}

// ---------------------------------------------------------------- kernel 3
// Transpose+convert: src[e][R][C] fp32 -> dst[e][C][R] bf16 (K-major for WMMA B).
__global__ void moe_transpose_cvt(const float* __restrict__ src,
                                  unsigned short* __restrict__ dst, int R, int C) {
  __shared__ float tile[32][33];
  const int e = blockIdx.z;
  const float* S = src + (size_t)e * R * C;
  unsigned short* D = dst + (size_t)e * R * C;
  const int c0 = blockIdx.x * 32, r0 = blockIdx.y * 32;
  const int tx = threadIdx.x & 31, ty = threadIdx.x >> 5;   // 32x8
#pragma unroll
  for (int i = 0; i < 32; i += 8)
    tile[ty + i][tx] = S[(size_t)(r0 + ty + i) * C + c0 + tx];
  __syncthreads();
#pragma unroll
  for (int i = 0; i < 32; i += 8)
    D[(size_t)(c0 + ty + i) * R + r0 + tx] = f2bf(tile[tx][ty + i]);
}

// ---------------------------------------------------------------- kernel 4
// One wave32 per token: router logits, noisy top-2, softmax gates, append.
__global__ void moe_router(const float* __restrict__ x, const float* __restrict__ noise,
                           const float* __restrict__ wr, const float* __restrict__ br,
                           const float* __restrict__ wn, const float* __restrict__ bn,
                           int* __restrict__ counts, int* __restrict__ toklist,
                           float* __restrict__ gatelist) {
  const int wave = threadIdx.x >> 5;
  const int lane = threadIdx.x & 31;
  const int t = blockIdx.x * NWAVES + wave;
  if (t >= T_TOK) return;

  float ar[NE], an[NE];
#pragma unroll
  for (int e = 0; e < NE; e++) { ar[e] = 0.f; an[e] = 0.f; }

  const float* xr = x + (size_t)t * DIN;
  for (int c = lane; c < DIN; c += 32) {
    const float xv = xr[c];
    const float* wrr = wr + c * NE;
    const float* wnr = wn + c * NE;
#pragma unroll
    for (int e = 0; e < NE; e++) { ar[e] += xv * wrr[e]; an[e] += xv * wnr[e]; }
  }
#pragma unroll
  for (int e = 0; e < NE; e++) {
#pragma unroll
    for (int m = 16; m >= 1; m >>= 1) {
      ar[e] += __shfl_xor(ar[e], m, 32);
      an[e] += __shfl_xor(an[e], m, 32);
    }
  }
  if (lane == 0) {
    float nz[NE];
    const float* no = noise + (size_t)t * NE;
#pragma unroll
    for (int e = 0; e < NE; e++) {
      float lg = ar[e] + br[e];
      float nl = an[e] + bn[e];
      float sp = (nl > 20.f) ? nl : log1pf(expf(nl));   // softplus
      nz[e] = lg + no[e] * sp;
    }
    int i1 = 0;
#pragma unroll
    for (int e = 1; e < NE; e++) if (nz[e] > nz[i1]) i1 = e;
    int i2 = (i1 == 0) ? 1 : 0;
#pragma unroll
    for (int e = 0; e < NE; e++) if (e != i1 && nz[e] > nz[i2]) i2 = e;
    float e2 = expf(nz[i2] - nz[i1]);
    float inv = 1.f / (1.f + e2);
    int p1 = atomicAdd(&counts[i1], 1);
    toklist[i1 * T_TOK + p1] = t;  gatelist[i1 * T_TOK + p1] = inv;
    int p2 = atomicAdd(&counts[i2], 1);
    toklist[i2 * T_TOK + p2] = t;  gatelist[i2 * T_TOK + p2] = e2 * inv;
  }
}

// ---------------------------------------------------------------- kernel 5
// One block = 32 routed tokens of one expert; bf16 WMMA, 2x2 register tiling.
// LDS: hs[32][DHID] bf16 (131072B) | ys[32][DIN] f32 (163840B, unions Xs bf16) | tail
#define LDS_HS   (TILE_M * DHID * 2)
#define LDS_YS   (TILE_M * DIN * 4)
#define LDS_TAIL 512
#define LDS_TOTAL (LDS_HS + LDS_YS + LDS_TAIL)

__global__ void moe_expert_ffn(const float* __restrict__ x,
                               const unsigned short* __restrict__ xb,
                               const unsigned short* __restrict__ w1t,  // [E][DHID][DIN] bf16
                               const float* __restrict__ b1,
                               const unsigned short* __restrict__ w2t,  // [E][DIN][DHID] bf16
                               const float* __restrict__ b2,
                               const float* __restrict__ gamma, const float* __restrict__ beta,
                               const int* __restrict__ counts, const int* __restrict__ toklist,
                               const float* __restrict__ gatelist,
                               float* __restrict__ out) {
  extern __shared__ char smem[];
  unsigned short* hs = (unsigned short*)smem;
  float*          ys = (float*)(smem + LDS_HS);
  unsigned short* Xs = (unsigned short*)(smem + LDS_HS);   // union with ys
  char*         tail = smem + LDS_HS + LDS_YS;
  int*   s_tok  = (int*)tail;            // [32]
  float* s_gate = (float*)(tail + 128);  // [32]
  float* s_mu   = (float*)(tail + 256);  // [32]
  float* s_rs   = (float*)(tail + 384);  // [32]

  const int e = blockIdx.y;
  const int cnt = counts[e];
  const int tile0 = blockIdx.x * TILE_M;
  if (tile0 >= cnt) return;

  if (threadIdx.x < TILE_M) {   // exactly wave 0
    const int idx = tile0 + threadIdx.x;
    const int tk = (idx < cnt) ? toklist[e * T_TOK + idx] : 0;
    s_tok[threadIdx.x]  = tk;
    s_gate[threadIdx.x] = (idx < cnt) ? gatelist[e * T_TOK + idx] : 0.f;

#if __has_builtin(__builtin_amdgcn_tensor_load_to_lds)
    // TDM gather-mode load: 2 ops x 16 rows of DIN bf16 each, indices = token ids.
    {
      const unsigned lds0 = (unsigned)(uintptr_t)(void*)Xs;
      const unsigned long long ga = (unsigned long long)(uintptr_t)xb;
      v8i g1;
      g1[0] = (int)(1u << 16);                    // data_size = 2 bytes
      g1[1] = (int)((unsigned)DIN << 16);         // tensor_dim0[15:0]
      g1[2] = (int)((unsigned)T_TOK << 16);       // tensor_dim1[15:0]
      g1[3] = (int)((unsigned)DIN << 16);         // tile_dim0
      g1[4] = 16;                                 // tile_dim1 = #indices
      g1[5] = DIN;                                // tensor_dim0_stride
      g1[6] = 0; g1[7] = 0;
#pragma unroll
      for (int h = 0; h < 2; h++) {
        int d[8];
#pragma unroll
        for (int j = 0; j < 8; j++) {
          int t0 = __builtin_amdgcn_readlane(tk, h * 16 + 2 * j);
          int t1 = __builtin_amdgcn_readlane(tk, h * 16 + 2 * j + 1);
          d[j] = (t0 & 0xFFFF) | (t1 << 16);
        }
        v4i g2, g3;
        g2[0] = d[0]; g2[1] = d[1]; g2[2] = d[2]; g2[3] = d[3];
        g3[0] = d[4]; g3[1] = d[5]; g3[2] = d[6]; g3[3] = d[7];
        v4u g0;
        g0[0] = 1u | (1u << 31);                  // count=1, gather_mode, 16b idx
        g0[1] = lds0 + (unsigned)(h * 16 * DIN * 2);
        g0[2] = (unsigned)(ga & 0xFFFFFFFFull);
        g0[3] = (unsigned)((ga >> 32) & 0x1FFFFFFull) | (2u << 30);  // type=2
#if __clang_major__ >= 23
        v8i g4 = {};
        __builtin_amdgcn_tensor_load_to_lds(g0, g1, g2, g3, g4, 0);
#else
        __builtin_amdgcn_tensor_load_to_lds(g0, g1, g2, g3, 0);
#endif
      }
      __builtin_amdgcn_s_wait_tensorcnt(0);
    }
#endif
  }
  __syncthreads();
#if !__has_builtin(__builtin_amdgcn_tensor_load_to_lds)
  for (int i = threadIdx.x; i < TILE_M * DIN; i += THREADS) {
    int m = i / DIN, c = i - m * DIN;
    Xs[m * DIN + c] = xb[(size_t)s_tok[m] * DIN + c];
  }
  __syncthreads();
#endif

  const int wave = threadIdx.x >> 5;
  const int lane = threadIdx.x & 31;
  const int hi   = lane >> 4;
  const int ln   = lane & 15;

  // ---------------- GEMM1: h = relu(X[32xDIN] @ W1 + b1), 2x2 tiles ---------
  const unsigned short* W1 = w1t + (size_t)e * DIN * DHID;
  for (int p = 0; p < (DHID / 16) / (NWAVES * 2); p++) {   // 8 passes
    const int nt0 = (p * NWAVES + wave) * 2;
    const int n0 = nt0 * 16, n1 = n0 + 16;
    v8f a00 = {}, a01 = {}, a10 = {}, a11 = {};
    for (int k0 = 0; k0 < DIN; k0 += 32) {
      v16u fA0 = load_afrag(Xs, ln,      DIN, k0, hi);
      v16u fA1 = load_afrag(Xs, 16 + ln, DIN, k0, hi);
      v16u fB0 = load_bfrag(W1, n0 + ln, DIN, k0, hi);
      v16u fB1 = load_bfrag(W1, n1 + ln, DIN, k0, hi);
      a00 = wmma_bf16(fA0, fB0, a00);
      a01 = wmma_bf16(fA0, fB1, a01);
      a10 = wmma_bf16(fA1, fB0, a10);
      a11 = wmma_bf16(fA1, fB1, a11);
    }
    const float bia0 = b1[(size_t)e * DHID + n0 + ln];
    const float bia1 = b1[(size_t)e * DHID + n1 + ln];
#pragma unroll
    for (int r = 0; r < 8; r++) {
      const int m = r + hi * 8;
      float v00 = a00[r] + bia0, v01 = a01[r] + bia1;
      float v10 = a10[r] + bia0, v11 = a11[r] + bia1;
      hs[m * DHID + n0 + ln]        = f2bf(v00 > 0.f ? v00 : 0.f);
      hs[m * DHID + n1 + ln]        = f2bf(v01 > 0.f ? v01 : 0.f);
      hs[(16 + m) * DHID + n0 + ln] = f2bf(v10 > 0.f ? v10 : 0.f);
      hs[(16 + m) * DHID + n1 + ln] = f2bf(v11 > 0.f ? v11 : 0.f);
    }
  }
  __syncthreads();

  // ---------------- GEMM2: y = h[32xDHID] @ W2 + b2 + x, 2x2 tiles ----------
  const unsigned short* W2 = w2t + (size_t)e * DIN * DHID;
  for (int p = 0; p < (DIN / 16) / (NWAVES * 2); p++) {    // 5 passes
    const int nt0 = (p * NWAVES + wave) * 2;
    const int n0 = nt0 * 16, n1 = n0 + 16;
    v8f a00 = {}, a01 = {}, a10 = {}, a11 = {};
    for (int k0 = 0; k0 < DHID; k0 += 32) {
      v16u fA0 = load_afrag(hs, ln,      DHID, k0, hi);
      v16u fA1 = load_afrag(hs, 16 + ln, DHID, k0, hi);
      v16u fB0 = load_bfrag(W2, n0 + ln, DHID, k0, hi);
      v16u fB1 = load_bfrag(W2, n1 + ln, DHID, k0, hi);
      a00 = wmma_bf16(fA0, fB0, a00);
      a01 = wmma_bf16(fA0, fB1, a01);
      a10 = wmma_bf16(fA1, fB0, a10);
      a11 = wmma_bf16(fA1, fB1, a11);
    }
    const float bia0 = b2[(size_t)e * DIN + n0 + ln];
    const float bia1 = b2[(size_t)e * DIN + n1 + ln];
#pragma unroll
    for (int r = 0; r < 8; r++) {
      const int m = r + hi * 8;
      const size_t x0 = (size_t)s_tok[m] * DIN;
      const size_t x1 = (size_t)s_tok[16 + m] * DIN;
      ys[m * DIN + n0 + ln]        = a00[r] + bia0 + x[x0 + n0 + ln];
      ys[m * DIN + n1 + ln]        = a01[r] + bia1 + x[x0 + n1 + ln];
      ys[(16 + m) * DIN + n0 + ln] = a10[r] + bia0 + x[x1 + n0 + ln];
      ys[(16 + m) * DIN + n1 + ln] = a11[r] + bia1 + x[x1 + n1 + ln];
    }
  }
  __syncthreads();

  // ---------------- LayerNorm stats: 4 rows per wave --------------------
  for (int m = wave; m < TILE_M; m += NWAVES) {
    float s = 0.f, s2 = 0.f;
    for (int c = lane; c < DIN; c += 32) {
      float v = ys[m * DIN + c];
      s += v; s2 += v * v;
    }
#pragma unroll
    for (int msk = 16; msk >= 1; msk >>= 1) {
      s  += __shfl_xor(s,  msk, 32);
      s2 += __shfl_xor(s2, msk, 32);
    }
    if (lane == 0) {
      float mu  = s / (float)DIN;
      float var = s2 / (float)DIN - mu * mu;
      s_mu[m] = mu;
      s_rs[m] = rsqrtf(var + LN_EPS);
    }
  }
  __syncthreads();

  // ---------------- normalize, affine, gate, scatter-add --------------------
  for (int i = threadIdx.x; i < TILE_M * DIN; i += THREADS) {
    const int m = i / DIN, c = i - m * DIN;
    const float g = s_gate[m];
    if (g != 0.f) {
      float v = (ys[i] - s_mu[m]) * s_rs[m];
      v = v * gamma[(size_t)e * DIN + c] + beta[(size_t)e * DIN + c];
      atomicAdd(&out[(size_t)s_tok[m] * DIN + c], g * v);
    }
  }
}

// ---------------------------------------------------------------- launch
extern "C" void kernel_launch(void* const* d_in, const int* in_sizes, int n_in,
                              void* d_out, int out_size, void* d_ws, size_t ws_size,
                              hipStream_t stream) {
  (void)in_sizes; (void)n_in; (void)out_size; (void)ws_size;
  const float* x     = (const float*)d_in[0];
  const float* noise = (const float*)d_in[1];
  const float* wr    = (const float*)d_in[2];
  const float* br    = (const float*)d_in[3];
  const float* wn    = (const float*)d_in[4];
  const float* bn    = (const float*)d_in[5];
  const float* w1    = (const float*)d_in[6];
  const float* b1    = (const float*)d_in[7];
  const float* w2    = (const float*)d_in[8];
  const float* b2    = (const float*)d_in[9];
  const float* gamma = (const float*)d_in[10];
  const float* beta  = (const float*)d_in[11];
  float* out = (float*)d_out;

  // workspace layout (256B aligned blocks)
  char* ws = (char*)d_ws;
  const size_t OFF_TOK  = 256;
  const size_t OFF_GATE = OFF_TOK + (size_t)NE * T_TOK * 4;
  const size_t OFF_XB   = OFF_GATE + (size_t)NE * T_TOK * 4;
  const size_t OFF_W1T  = OFF_XB + (size_t)T_TOK * DIN * 2;
  const size_t OFF_W2T  = OFF_W1T + (size_t)NE * DIN * DHID * 2;
  int*            counts   = (int*)ws;
  int*            toklist  = (int*)(ws + OFF_TOK);
  float*          gatelist = (float*)(ws + OFF_GATE);
  unsigned short* xb       = (unsigned short*)(ws + OFF_XB);
  unsigned short* w1t      = (unsigned short*)(ws + OFF_W1T);
  unsigned short* w2t      = (unsigned short*)(ws + OFF_W2T);

  const int nx = T_TOK * DIN;
  moe_zero_init<<<(nx + 255) / 256, 256, 0, stream>>>(out, nx, counts);
  moe_cvt_x<<<(nx + 255) / 256, 256, 0, stream>>>(x, xb, nx);
  // w1[e]: [DIN][DHID] -> w1t[e]: [DHID][DIN];  w2[e]: [DHID][DIN] -> w2t[e]: [DIN][DHID]
  moe_transpose_cvt<<<dim3(DHID / 32, DIN / 32, NE), 256, 0, stream>>>(w1, w1t, DIN, DHID);
  moe_transpose_cvt<<<dim3(DIN / 32, DHID / 32, NE), 256, 0, stream>>>(w2, w2t, DHID, DIN);
  moe_router<<<T_TOK / NWAVES, THREADS, 0, stream>>>(x, noise, wr, br, wn, bn,
                                                     counts, toklist, gatelist);

  hipFuncSetAttribute((const void*)moe_expert_ffn,
                      hipFuncAttributeMaxDynamicSharedMemorySize, LDS_TOTAL);
  dim3 grid(T_TOK / TILE_M, NE);   // 256 tiles x 8 experts, inactive tiles exit early
  moe_expert_ffn<<<grid, THREADS, LDS_TOTAL, stream>>>(
      x, xb, w1t, b1, w2t, b2, gamma, beta, counts, toklist, gatelist, out);
}
